// SpatioTemporalGAT_32461362823923
// MI455X (gfx1250) — matrix operand order repaired and checked by reading, not compiled, not run on hardware
//
#include <hip/hip_runtime.h>

// SpatioTemporal GAT forward for MI455X (gfx1250, wave32, WMMA).
// N=4096 nodes, F=H=64, K=2 heads, O=32.
// k1: Wh = X @ Wgat^T  (bf16 WMMA)  + log2e-prescaled s_src/s_dst + Wh^T (bf16)
// k2: fused masked-softmax attention (flash-style, single pass):
//     unnormalized exp2 weights accumulated via WMMA, normalize+elu at end.
//     adj (64MB) is the only HBM-bound stream -> prefetched ahead.
// k3: LSTM gates (i,g,o; f dead since c0=0) + fc, all WMMA, LDS re-layout.

#define N_NODES 4096
#define FDIM 64
#define HDIM 64
#define NHEAD 2
#define ODIM 32
#define LOG2E 1.4426950408889634f

typedef __attribute__((ext_vector_type(16))) __bf16 v16bf;
typedef __attribute__((ext_vector_type(16))) unsigned short v16u;
typedef __attribute__((ext_vector_type(8))) float v8f;

static __device__ __forceinline__ unsigned short f2bf(float f) {
  // native f32->bf16 (RNE); backend picks the hw convert op
  return __builtin_bit_cast(unsigned short, static_cast<__bf16>(f));
}

static __device__ __forceinline__ v8f vzero() {
  v8f z = {0.f, 0.f, 0.f, 0.f, 0.f, 0.f, 0.f, 0.f};
  return z;
}

static __device__ __forceinline__ v8f wmma_bf16(v16u a, v16u b, v8f c) {
  return __builtin_amdgcn_wmma_f32_16x16x32_bf16(
      false, __builtin_bit_cast(v16bf, a),
      false, __builtin_bit_cast(v16bf, b),
      (short)0, c, false, false);
}

template <int M>
static __device__ __forceinline__ float swz_xor(float v) {
  return __builtin_bit_cast(
      float, __builtin_amdgcn_ds_swizzle(__builtin_bit_cast(int, v),
                                         (M << 10) | 0x1f));
}

static __device__ __forceinline__ float lane_bcast(float v, int srcl) {
  return __builtin_bit_cast(
      float, __builtin_amdgcn_ds_bpermute(srcl << 2, __builtin_bit_cast(int, v)));
}

// A-matrix (16-bit, 16x32) element e -> K offset, per lane half.
static __device__ __forceinline__ int a_koff(int e, int hi) {
  return e + (hi ? 16 : 0);
}

static __device__ __forceinline__ float sigmoidf_(float x) {
  return 1.f / (1.f + __expf(-x));
}

// ---------------------------------------------------------------- kernel 1
__global__ __launch_bounds__(128) void k1_wh(
    const float* __restrict__ X, const float* __restrict__ Wg,
    const float* __restrict__ a_src, const float* __restrict__ a_dst,
    unsigned short* __restrict__ WhT, float* __restrict__ s_src,
    float* __restrict__ s_dst) {
  const int w = (blockIdx.x * blockDim.x + threadIdx.x) >> 5;
  const int lane = threadIdx.x & 31;
  const int head = w & 1;
  const int n0 = (w >> 1) * 16;
  const int hi = lane >> 4;
  const int lc = lane & 15;
  const int kstart = hi ? 16 : 0;
  const int b8 = hi ? 8 : 0;

  // A: X tile 16x64 (2 K-chunks of 32), bf16 in A layout
  v16u A[2];
#pragma unroll
  for (int c = 0; c < 2; ++c)
#pragma unroll
    for (int e = 0; e < 16; ++e)
      A[c][e] = f2bf(X[(n0 + lc) * FDIM + 32 * c + a_koff(e, hi)]);

  v8f acc[4];
#pragma unroll
  for (int t = 0; t < 4; ++t) acc[t] = vzero();

#pragma unroll
  for (int c = 0; c < 2; ++c) {
#pragma unroll
    for (int t = 0; t < 4; ++t) {
      // B[k=f, n=h] = Wgat[head][h][f]
      const float* wrow =
          Wg + head * HDIM * FDIM + (16 * t + lc) * FDIM + 32 * c + kstart;
      v16u B;
#pragma unroll
      for (int e = 0; e < 16; ++e) B[e] = f2bf(wrow[e]);
      acc[t] = wmma_bf16(A[c], B, acc[t]);
    }
  }

  // s_src[n] = log2e * Wh[n,:].a_src ; s_dst likewise (prescaled so the
  // attention kernel can use a single v_exp_f32 via exp2).
  float ps[8], pd[8];
#pragma unroll
  for (int r = 0; r < 8; ++r) { ps[r] = 0.f; pd[r] = 0.f; }
#pragma unroll
  for (int t = 0; t < 4; ++t) {
    float as = a_src[head * HDIM + 16 * t + lc];
    float ad = a_dst[head * HDIM + 16 * t + lc];
#pragma unroll
    for (int r = 0; r < 8; ++r) {
      ps[r] += acc[t][r] * as;
      pd[r] += acc[t][r] * ad;
    }
  }
#pragma unroll
  for (int r = 0; r < 8; ++r) {  // reduce across 16 lanes of each half
    ps[r] += swz_xor<1>(ps[r]); ps[r] += swz_xor<2>(ps[r]);
    ps[r] += swz_xor<4>(ps[r]); ps[r] += swz_xor<8>(ps[r]);
    pd[r] += swz_xor<1>(pd[r]); pd[r] += swz_xor<2>(pd[r]);
    pd[r] += swz_xor<4>(pd[r]); pd[r] += swz_xor<8>(pd[r]);
  }
  if (lc == 0) {
#pragma unroll
    for (int r = 0; r < 8; ++r) {
      s_src[head * N_NODES + n0 + b8 + r] = ps[r] * LOG2E;
      s_dst[head * N_NODES + n0 + b8 + r] = pd[r] * LOG2E;
    }
  }

  // WhT[head][h][n] bf16: lane stores 8 consecutive n as one b128
#pragma unroll
  for (int t = 0; t < 4; ++t) {
    uint4 u;
    u.x = (unsigned)f2bf(acc[t][0]) | ((unsigned)f2bf(acc[t][1]) << 16);
    u.y = (unsigned)f2bf(acc[t][2]) | ((unsigned)f2bf(acc[t][3]) << 16);
    u.z = (unsigned)f2bf(acc[t][4]) | ((unsigned)f2bf(acc[t][5]) << 16);
    u.w = (unsigned)f2bf(acc[t][6]) | ((unsigned)f2bf(acc[t][7]) << 16);
    *(uint4*)(WhT + (((head * HDIM + 16 * t + lc) << 12) + n0 + b8)) = u;
  }
}

// ---------------------------------------------------------------- kernel 2
__global__ __launch_bounds__(128) void k2_attn(
    const int* __restrict__ adj, const unsigned short* __restrict__ WhT,
    const float* __restrict__ s_src, const float* __restrict__ s_dst,
    float* __restrict__ hbuf) {
  const int w = (blockIdx.x * blockDim.x + threadIdx.x) >> 5;
  const int lane = threadIdx.x & 31;
  const int head = w & 1;
  const int p0 = (w >> 1) * 16;
  const int hi = lane >> 4;
  const int lc = lane & 15;
  const int kstart = hi ? 16 : 0;
  const int g8 = hi ? 8 : 0;

  const float sdst = s_dst[head * N_NODES + p0 + lc];   // my A-row's dst term
  const int* adjrow = adj + (size_t)(p0 + lc) * N_NODES;
  const float* ssp = s_src + head * N_NODES;
  const unsigned short* whb = WhT + ((head * HDIM) << 12);

  v8f acc0 = vzero(), acc1 = vzero(), acc2 = vzero(), acc3 = vzero();
  float ssum = 0.f;   // unnormalized exp sum over this lane's K slots (row lc)

  for (int q0 = 0; q0 < N_NODES; q0 += 32) {
    // prefetch the adjacency stream (only HBM-bound operand) ~16 iters ahead
    if (q0 + 512 < N_NODES)
      __builtin_prefetch(adjrow + q0 + 512 + g8, 0, 0);

    float sv[16];
    int av[16];
    const float4* s4 = (const float4*)(ssp + q0 + kstart);
    const int4* a4 = (const int4*)(adjrow + q0 + kstart);
    *(float4*)&sv[0] = s4[0];  *(float4*)&sv[4] = s4[1];   // K +0..7
    *(float4*)&sv[8] = s4[2];  *(float4*)&sv[12] = s4[3];  // K +8..15
    *(int4*)&av[0] = a4[0];    *(int4*)&av[4] = a4[1];
    *(int4*)&av[8] = a4[2];    *(int4*)&av[12] = a4[3];

    v16u A;
#pragma unroll
    for (int e = 0; e < 16; ++e) {
      float x = sdst + sv[e];                 // already scaled by log2(e)
      x = x > 0.f ? x : 0.2f * x;             // lrelu commutes with +scale
      float p = av[e] > 0 ? __builtin_amdgcn_exp2f(x) : 0.f;  // mask -> 0
      ssum += p;
      A[e] = f2bf(p);
    }
    // B: WhT[head][h][q0+kstart .. +15], 16 contiguous bf16 per lane
    const unsigned short* wb = whb + q0 + kstart;
    v16u B0 = *(const v16u*)(wb + ((0 + lc) << 12));
    v16u B1 = *(const v16u*)(wb + ((16 + lc) << 12));
    v16u B2 = *(const v16u*)(wb + ((32 + lc) << 12));
    v16u B3 = *(const v16u*)(wb + ((48 + lc) << 12));
    acc0 = wmma_bf16(A, B0, acc0);
    acc1 = wmma_bf16(A, B1, acc1);
    acc2 = wmma_bf16(A, B2, acc2);
    acc3 = wmma_bf16(A, B3, acc3);
  }

  float rowsum = ssum + swz_xor<16>(ssum);  // every lane: sum for row lc
#pragma unroll
  for (int r = 0; r < 8; ++r) {
    float inv = 1.f / lane_bcast(rowsum, g8 + r);  // sum for D-row g8+r
    float o0 = acc0[r] * inv, o1 = acc1[r] * inv;
    float o2 = acc2[r] * inv, o3 = acc3[r] * inv;
    o0 = o0 > 0.f ? o0 : __expf(o0) - 1.f;   // elu
    o1 = o1 > 0.f ? o1 : __expf(o1) - 1.f;
    o2 = o2 > 0.f ? o2 : __expf(o2) - 1.f;
    o3 = o3 > 0.f ? o3 : __expf(o3) - 1.f;
    float* hr =
        hbuf + (size_t)(p0 + g8 + r) * (NHEAD * HDIM) + head * HDIM + lc;
    hr[0] = o0; hr[16] = o1; hr[32] = o2; hr[48] = o3;
  }
}

// ---------------------------------------------------------------- kernel 3
__global__ __launch_bounds__(128) void k3_lstm_fc(
    const float* __restrict__ hbuf, const float* __restrict__ W_ih,
    const float* __restrict__ b_ih, const float* __restrict__ b_hh,
    const float* __restrict__ fc_W, const float* __restrict__ fc_b,
    float* __restrict__ out) {
  __shared__ float ht_s[4][16 * HDIM];
  const int wi = threadIdx.x >> 5;
  const int w = (blockIdx.x * blockDim.x + threadIdx.x) >> 5;
  const int lane = threadIdx.x & 31;
  const int n0 = w * 16;
  const int hi = lane >> 4;
  const int lc = lane & 15;
  const int kstart = hi ? 16 : 0;
  const int g8 = hi ? 8 : 0;

  // A: h tile 16x128 (4 K-chunks of 32)
  v16u Ag[4];
  const float* hrow = hbuf + (size_t)(n0 + lc) * (NHEAD * HDIM);
#pragma unroll
  for (int c = 0; c < 4; ++c)
#pragma unroll
    for (int e = 0; e < 16; ++e)
      Ag[c][e] = f2bf(hrow[32 * c + a_koff(e, hi)]);

#pragma unroll
  for (int jb = 0; jb < 4; ++jb) {  // 16-wide column block within H
    v8f ai = vzero(), ag = vzero(), ao = vzero();
    const int ci = 0 * HDIM + 16 * jb + lc;  // i gate column
    const int cg = 2 * HDIM + 16 * jb + lc;  // g gate column (f is dead)
    const int co = 3 * HDIM + 16 * jb + lc;  // o gate column
#pragma unroll
    for (int c = 0; c < 4; ++c) {
      const float* pi = W_ih + (size_t)ci * 128 + 32 * c + kstart;
      const float* pg = W_ih + (size_t)cg * 128 + 32 * c + kstart;
      const float* po = W_ih + (size_t)co * 128 + 32 * c + kstart;
      v16u Bi, Bg, Bo;
#pragma unroll
      for (int e = 0; e < 16; ++e) {
        Bi[e] = f2bf(pi[e]); Bg[e] = f2bf(pg[e]); Bo[e] = f2bf(po[e]);
      }
      ai = wmma_bf16(Ag[c], Bi, ai);
      ag = wmma_bf16(Ag[c], Bg, ag);
      ao = wmma_bf16(Ag[c], Bo, ao);
    }
    const float bi_ = b_ih[ci] + b_hh[ci];
    const float bg_ = b_ih[cg] + b_hh[cg];
    const float bo_ = b_ih[co] + b_hh[co];
#pragma unroll
    for (int r = 0; r < 8; ++r) {
      float iv = sigmoidf_(ai[r] + bi_);
      float gv = tanhf(ag[r] + bg_);
      float ov = sigmoidf_(ao[r] + bo_);
      float cv = iv * gv;                       // f*c0 == 0
      ht_s[wi][(g8 + r) * HDIM + 16 * jb + lc] = ov * tanhf(cv);
    }
  }
  __syncthreads();

  // A: ht tile 16x64 from LDS (2 K-chunks)
  v16u Ah[2];
#pragma unroll
  for (int c = 0; c < 2; ++c)
#pragma unroll
    for (int e = 0; e < 16; ++e)
      Ah[c][e] = f2bf(ht_s[wi][lc * HDIM + 32 * c + a_koff(e, hi)]);

#pragma unroll
  for (int t = 0; t < 2; ++t) {  // two 16-wide output-column tiles (O=32)
    v8f acc = vzero();
#pragma unroll
    for (int c = 0; c < 2; ++c) {
      const float* fr = fc_W + (size_t)(16 * t + lc) * HDIM + 32 * c + kstart;
      v16u B;
#pragma unroll
      for (int e = 0; e < 16; ++e) B[e] = f2bf(fr[e]);
      acc = wmma_bf16(Ah[c], B, acc);
    }
    const float bb = fc_b[16 * t + lc];
#pragma unroll
    for (int r = 0; r < 8; ++r)
      out[(size_t)(n0 + g8 + r) * ODIM + 16 * t + lc] = acc[r] + bb;
  }
}

// ---------------------------------------------------------------- launch
extern "C" void kernel_launch(void* const* d_in, const int* in_sizes, int n_in,
                              void* d_out, int out_size, void* d_ws,
                              size_t ws_size, hipStream_t stream) {
  (void)in_sizes; (void)n_in; (void)out_size; (void)ws_size;
  const float* X    = (const float*)d_in[0];
  const int*   adj  = (const int*)d_in[1];
  // d_in[2] temporal_features: unused by the reference forward pass
  const float* Wg   = (const float*)d_in[3];
  const float* a_s  = (const float*)d_in[4];
  const float* a_d  = (const float*)d_in[5];
  const float* W_ih = (const float*)d_in[6];
  // d_in[7] W_hh: dead (h0 == 0)
  const float* b_ih = (const float*)d_in[8];
  const float* b_hh = (const float*)d_in[9];
  const float* fcW  = (const float*)d_in[10];
  const float* fcb  = (const float*)d_in[11];
  float* out = (float*)d_out;

  // workspace layout (~3.1 MB)
  unsigned short* WhT = (unsigned short*)d_ws;            // [2][64][4096] bf16, 1MB
  float* s_src = (float*)((char*)d_ws + (1u << 20));      // [2][4096] (x log2e)
  float* s_dst = s_src + NHEAD * N_NODES;                 // [2][4096] (x log2e)
  float* hbuf  = s_dst + NHEAD * N_NODES;                 // [4096][128] f32, 2MB

  k1_wh<<<128, 128, 0, stream>>>(X, Wg, a_s, a_d, WhT, s_src, s_dst);
  k2_attn<<<128, 128, 0, stream>>>(adj, WhT, s_src, s_dst, hbuf);
  k3_lstm_fc<<<64, 128, 0, stream>>>(hbuf, W_ih, b_ih, b_hh, fcW, fcb, out);
}
